// gin_tuple2_Net_67508295958860
// MI455X (gfx1250) — matrix operand order repaired and checked by reading, not compiled, not run on hardware
//
#include <hip/hip_runtime.h>
#include <hip/hip_bf16.h>

typedef __bf16 bf16_t;
typedef __attribute__((ext_vector_type(16))) __bf16 v16bf;
typedef __attribute__((ext_vector_type(8)))  float  v8f;

#define N_NODES_C   100000
#define N_EDGES_C   1000000
#define NUM_GRAPHS_C 128
#define DH          128

// ---------------------------------------------------------------------------
// WMMA bf16 16x16x32 layout helpers (wave32)
//   A (16xK): lane l -> row m=l&15, half hh=l>>4; element j of the v16bf holds
//             k = kb*32 + (j>>3)*16 + hh*8 + (j&7)
//   B (Kx16): lane l -> col n=l&15, half hh=l>>4; element j holds
//             k = kb*32 + (j>>3)*16 + hh*8 + (j&7)
//   C/D f32 : VGPR r -> row m = r + 8*(lane>>4), col n = lane&15
// We store A tiles in LDS (and B weights in global) pre-permuted so each
// lane's fragment is one contiguous 32-byte vector load.
// ---------------------------------------------------------------------------
__device__ __forceinline__ int aperm_off(int k) {
    int kb = k >> 5;
    int r  = k & 31;
    int hh = (r >> 3) & 1;
    int j  = ((r >> 4) << 3) | (r & 7);
    return (kb * 2 + hh) * 16 + j;
}

__device__ __forceinline__ void atomAddF(float* p, float v) {
    unsafeAtomicAdd(p, v);   // hardware global_atomic_add_f32
}

// ---------------------------------------------------------------------------
// Convert fp32 weight [K x 128] into B-fragment-permuted bf16:
//   dst[((kb*2 + hh)*128 + n)*16 + j]  with k = kb*32 + (j>>3)*16 + hh*8 + (j&7)
// so lane(col n) reads one v16bf per (kb,hh).
// ---------------------------------------------------------------------------
__global__ void convert_weight_kernel(const float* __restrict__ src,
                                      bf16_t* __restrict__ dst, int K) {
    int i = blockIdx.x * blockDim.x + threadIdx.x;
    int n = K * DH;
    if (i >= n) return;
    int k   = i / DH;
    int col = i - k * DH;
    int kb = k >> 5;
    int hh = (k >> 3) & 1;
    int j  = (((k >> 4) & 1) << 3) | (k & 7);
    dst[(((long)kb * 2 + hh) * DH + col) * 16 + j] = (bf16_t)src[i];
}

// ---------------------------------------------------------------------------
// Edge scatter: agg[dst] += xin[src], vectorized float4 + f32 atomics.
// ei is [2, nE] : row 0 = source, row 1 = destination.
// ---------------------------------------------------------------------------
__global__ void scatter_add_kernel(const float* __restrict__ xin, int ld,
                                   const int* __restrict__ ei,
                                   float* __restrict__ agg,
                                   int nE, int dq /* = d/4 */) {
    long gid = (long)blockIdx.x * blockDim.x + threadIdx.x;
    long total = (long)nE * dq;
    if (gid >= total) return;
    int e = (int)(gid % nE);
    int c = (int)(gid / nE) * 4;
    int src = ei[e];
    int dst = ei[nE + e];
    const float4 v = *(const float4*)(xin + (long)src * ld + c);
    float* p = agg + (long)dst * DH + c;
    atomAddF(p + 0, v.x);
    atomAddF(p + 1, v.y);
    atomAddF(p + 2, v.z);
    atomAddF(p + 3, v.w);
}

// ---------------------------------------------------------------------------
// Fused 2-layer MLP on node features using bf16 WMMA, f32 accumulation.
//   CAT=0:  h = in0[row, 0:K1](ld0) + in1[row, 0:K1](ld 128)        (GIN agg)
//   CAT=1:  h = concat(in0[row,0:128], in1[row,0:128])   (K1 must be 256)
//   out[row, 0:128] = f2( f1relu( h @ W1 + b1 ) @ W2 + b2 )
// Block = 256 threads (8 waves); block owns 16 rows; wave w owns cols [16w,16w+16).
// ---------------------------------------------------------------------------
template <int K1, bool CAT, bool RELU_OUT>
__global__ __launch_bounds__(256)
void gin_mlp_kernel(const float* __restrict__ in0, int ld0,
                    const float* __restrict__ in1,
                    const bf16_t* __restrict__ w1, const float* __restrict__ b1,
                    const bf16_t* __restrict__ w2, const float* __restrict__ b2,
                    float* __restrict__ out, int nRows) {
    __shared__ alignas(32) bf16_t aP[16 * (K1 + 16)];   // A tile, WMMA-permuted (+pad)
    __shared__ alignas(32) bf16_t tP[16 * (DH + 16)];   // intermediate tile, same layout

    const int tid     = threadIdx.x;
    const int lane    = tid & 31;
    const int wv      = tid >> 5;           // 0..7 -> output column tile
    const int rowBase = blockIdx.x * 16;

    // ---- stage input tile (16 x K1) into LDS as bf16, A-permuted ----
    for (int idx = tid; idx < 16 * K1; idx += 256) {
        int m = idx / K1;
        int k = idx - m * K1;
        int row = rowBase + m;
        float v = 0.f;
        if (row < nRows) {
            if (CAT) {
                v = (k < DH) ? in0[(long)row * DH + k]
                             : in1[(long)row * DH + (k - DH)];
            } else {
                v = in0[(long)row * ld0 + k] + in1[(long)row * DH + k];
            }
        }
        aP[m * (K1 + 16) + aperm_off(k)] = (bf16_t)v;
    }
    __syncthreads();

    const int mC   = lane & 15;     // A row / C col within tile
    const int hh   = lane >> 4;     // half-wave
    const int nCol = wv * 16 + mC;  // global output column for B/C/D

    const v16bf* w1v = (const v16bf*)w1;
    const v16bf* w2v = (const v16bf*)w2;

    // ---- GEMM1: (16 x K1) @ (K1 x 128) ----
    v8f acc = {};
#pragma unroll
    for (int kb = 0; kb < K1 / 32; ++kb) {
        v16bf a = *(const v16bf*)&aP[mC * (K1 + 16) + (kb * 2 + hh) * 16];
        v16bf b = w1v[(kb * 2 + hh) * DH + nCol];
        acc = __builtin_amdgcn_wmma_f32_16x16x32_bf16(
            false, a, false, b, (short)0, acc, false, false);
    }
    // bias + ReLU, write intermediate tile to LDS in A layout (n becomes k)
    {
        float bias = b1[nCol];
        int toff = aperm_off(nCol);
#pragma unroll
        for (int r = 0; r < 8; ++r) {
            float v = acc[r] + bias;
            v = v > 0.f ? v : 0.f;
            int mRow = r + hh * 8;
            tP[mRow * (DH + 16) + toff] = (bf16_t)v;
        }
    }
    __syncthreads();

    // ---- GEMM2: (16 x 128) @ (128 x 128) ----
    v8f acc2 = {};
#pragma unroll
    for (int kb = 0; kb < DH / 32; ++kb) {
        v16bf a = *(const v16bf*)&tP[mC * (DH + 16) + (kb * 2 + hh) * 16];
        v16bf b = w2v[(kb * 2 + hh) * DH + nCol];
        acc2 = __builtin_amdgcn_wmma_f32_16x16x32_bf16(
            false, a, false, b, (short)0, acc2, false, false);
    }
    // bias (+ optional ReLU) and store
    {
        float bias = b2[nCol];
#pragma unroll
        for (int r = 0; r < 8; ++r) {
            float v = acc2[r] + bias;
            if (RELU_OUT) v = v > 0.f ? v : 0.f;
            int row = rowBase + r + hh * 8;
            if (row < nRows) out[(long)row * DH + nCol] = v;
        }
    }
}

// ---------------------------------------------------------------------------
// Global add pool: g[batch[i]] += h[i]
// ---------------------------------------------------------------------------
__global__ void pool_kernel(const float* __restrict__ h,
                            const int* __restrict__ batch,
                            float* __restrict__ g, int nRows) {
    long gid = (long)blockIdx.x * blockDim.x + threadIdx.x;
    long total = (long)nRows * (DH / 4);
    if (gid >= total) return;
    int i = (int)(gid % nRows);
    int c = (int)(gid / nRows) * 4;
    int grp = batch[i];
    const float4 v = *(const float4*)(h + (long)i * DH + c);
    float* p = g + (long)grp * DH + c;
    atomAddF(p + 0, v.x);
    atomAddF(p + 1, v.y);
    atomAddF(p + 2, v.z);
    atomAddF(p + 3, v.w);
}

// ---------------------------------------------------------------------------
// Final head: out[g] = dot(gpool[g], lin_w) + lin_b
// ---------------------------------------------------------------------------
__global__ void head_kernel(const float* __restrict__ g,
                            const float* __restrict__ w,
                            const float* __restrict__ b,
                            float* __restrict__ out) {
    int gi = blockIdx.x * blockDim.x + threadIdx.x;
    if (gi >= NUM_GRAPHS_C) return;
    float s = b[0];
#pragma unroll 8
    for (int k = 0; k < DH; ++k) s += g[(long)gi * DH + k] * w[k];
    out[gi] = s;
}

// ---------------------------------------------------------------------------
extern "C" void kernel_launch(void* const* d_in, const int* in_sizes, int n_in,
                              void* d_out, int out_size, void* d_ws, size_t ws_size,
                              hipStream_t stream) {
    (void)in_sizes; (void)n_in; (void)out_size; (void)ws_size;

    const float* x      = (const float*)d_in[0];
    const int*   ei0    = (const int*)d_in[1];
    const int*   ei1    = (const int*)d_in[2];
    const int*   batch  = (const int*)d_in[3];
    const float* c11_w1 = (const float*)d_in[4];
    const float* c11_b1 = (const float*)d_in[5];
    const float* c11_w2 = (const float*)d_in[6];
    const float* c11_b2 = (const float*)d_in[7];
    const float* c12_w1 = (const float*)d_in[8];
    const float* c12_b1 = (const float*)d_in[9];
    const float* c12_w2 = (const float*)d_in[10];
    const float* c12_b2 = (const float*)d_in[11];
    const float* m1_w1  = (const float*)d_in[12];
    const float* m1_b1  = (const float*)d_in[13];
    const float* m1_w2  = (const float*)d_in[14];
    const float* m1_b2  = (const float*)d_in[15];
    const float* c21_w1 = (const float*)d_in[16];
    const float* c21_b1 = (const float*)d_in[17];
    const float* c21_w2 = (const float*)d_in[18];
    const float* c21_b2 = (const float*)d_in[19];
    const float* c22_w1 = (const float*)d_in[20];
    const float* c22_b1 = (const float*)d_in[21];
    const float* c22_w2 = (const float*)d_in[22];
    const float* c22_b2 = (const float*)d_in[23];
    const float* m2_w1  = (const float*)d_in[24];
    const float* m2_b1  = (const float*)d_in[25];
    const float* m2_w2  = (const float*)d_in[26];
    const float* m2_b2  = (const float*)d_in[27];
    const float* lin_w  = (const float*)d_in[28];
    const float* lin_b  = (const float*)d_in[29];
    float* outp = (float*)d_out;

    // workspace carve-up
    float* bufA = (float*)d_ws;                               // N x 128
    float* bufB = bufA + (size_t)N_NODES_C * DH;              // N x 128
    float* bufH = bufB + (size_t)N_NODES_C * DH;              // N x 128
    float* gbuf = bufH + (size_t)N_NODES_C * DH;              // 128 x 128
    bf16_t* wcur = (bf16_t*)(gbuf + (size_t)NUM_GRAPHS_C * DH);

    auto cvtW = [&](const float* src, int K) -> bf16_t* {
        bf16_t* dst = wcur;
        int n = K * DH;
        wcur += n;
        convert_weight_kernel<<<(n + 255) / 256, 256, 0, stream>>>(src, dst, K);
        return dst;
    };

    // bf16 weight conversion (permuted for WMMA B fragments)
    bf16_t* pc11w1 = cvtW(c11_w1, 64);
    bf16_t* pc11w2 = cvtW(c11_w2, 128);
    bf16_t* pc12w1 = cvtW(c12_w1, 64);
    bf16_t* pc12w2 = cvtW(c12_w2, 128);
    bf16_t* pm1w1  = cvtW(m1_w1, 256);
    bf16_t* pm1w2  = cvtW(m1_w2, 128);
    bf16_t* pc21w1 = cvtW(c21_w1, 128);
    bf16_t* pc21w2 = cvtW(c21_w2, 128);
    bf16_t* pc22w1 = cvtW(c22_w1, 128);
    bf16_t* pc22w2 = cvtW(c22_w2, 128);
    bf16_t* pm2w1  = cvtW(m2_w1, 256);
    bf16_t* pm2w2  = cvtW(m2_w2, 128);

    const size_t nodeBytes = (size_t)N_NODES_C * DH * sizeof(float);
    const int gridM = (N_NODES_C + 15) / 16;   // 6250

    // ---------------- layer 1 ----------------
    hipMemsetAsync(bufA, 0, nodeBytes, stream);
    hipMemsetAsync(bufB, 0, nodeBytes, stream);
    {
        long tot = (long)N_EDGES_C * 16;       // d=64 -> 16 float4 chunks
        int blocks = (int)((tot + 255) / 256);
        scatter_add_kernel<<<blocks, 256, 0, stream>>>(x, 64, ei0, bufA, N_EDGES_C, 16);
        scatter_add_kernel<<<blocks, 256, 0, stream>>>(x, 64, ei1, bufB, N_EDGES_C, 16);
    }
    // conv1a/conv1b (no output relu): in-place on agg buffers (each block owns its rows)
    gin_mlp_kernel<64, false, false><<<gridM, 256, 0, stream>>>(
        x, 64, bufA, pc11w1, c11_b1, pc11w2, c11_b2, bufA, N_NODES_C);
    gin_mlp_kernel<64, false, false><<<gridM, 256, 0, stream>>>(
        x, 64, bufB, pc12w1, c12_b1, pc12w2, c12_b2, bufB, N_NODES_C);
    // concat MLP 1 -> bufH
    gin_mlp_kernel<256, true, false><<<gridM, 256, 0, stream>>>(
        bufA, DH, bufB, pm1w1, m1_b1, pm1w2, m1_b2, bufH, N_NODES_C);

    // ---------------- layer 2 ----------------
    hipMemsetAsync(bufA, 0, nodeBytes, stream);
    hipMemsetAsync(bufB, 0, nodeBytes, stream);
    {
        long tot = (long)N_EDGES_C * 32;       // d=128 -> 32 float4 chunks
        int blocks = (int)((tot + 255) / 256);
        scatter_add_kernel<<<blocks, 256, 0, stream>>>(bufH, DH, ei0, bufA, N_EDGES_C, 32);
        scatter_add_kernel<<<blocks, 256, 0, stream>>>(bufH, DH, ei1, bufB, N_EDGES_C, 32);
    }
    // conv2a/conv2b WITH output relu
    gin_mlp_kernel<128, false, true><<<gridM, 256, 0, stream>>>(
        bufH, DH, bufA, pc21w1, c21_b1, pc21w2, c21_b2, bufA, N_NODES_C);
    gin_mlp_kernel<128, false, true><<<gridM, 256, 0, stream>>>(
        bufH, DH, bufB, pc22w1, c22_b1, pc22w2, c22_b2, bufB, N_NODES_C);
    // concat MLP 2 -> bufH
    gin_mlp_kernel<256, true, false><<<gridM, 256, 0, stream>>>(
        bufA, DH, bufB, pm2w1, m2_b1, pm2w2, m2_b2, bufH, N_NODES_C);

    // ---------------- pool + head ----------------
    hipMemsetAsync(gbuf, 0, (size_t)NUM_GRAPHS_C * DH * sizeof(float), stream);
    {
        long tot = (long)N_NODES_C * 32;
        int blocks = (int)((tot + 255) / 256);
        pool_kernel<<<blocks, 256, 0, stream>>>(bufH, batch, gbuf, N_NODES_C);
    }
    head_kernel<<<1, NUM_GRAPHS_C, 0, stream>>>(gbuf, lin_w, lin_b, outp);
}
